// ScaleEncoder_87247965651442
// MI455X (gfx1250) — matrix operand order repaired and checked by reading, not compiled, not run on hardware
//
#include <hip/hip_runtime.h>
#include <hip/hip_bf16.h>
#include <math.h>

// ---------------------------------------------------------------------------
// Types for WMMA fragments (gfx1250, wave32)
// ---------------------------------------------------------------------------
typedef __attribute__((ext_vector_type(16))) _Float16 v16h;
typedef __attribute__((ext_vector_type(8)))  _Float16 v8h;
typedef __attribute__((ext_vector_type(8)))  float    v8f;

// Problem constants
#define BSZ 4
#define CCH 128
#define DCH 128
#define NHD 4
#define NLV 2
#define NPT 8
#define H0 64
#define W0 96
#define H1 32
#define W1 48
#define N0 (H0*W0)      // 6144
#define N1 (H1*W1)      // 1536
#define NTOT (N0+N1)    // 7680

// ---------------------------------------------------------------------------
// Generic weight convert:  dst[r*cols+c] = (trans ? src[c*rows+r] : src[r*cols+c])
// ---------------------------------------------------------------------------
__global__ void k_convert_w(const float* __restrict__ src, _Float16* __restrict__ dst,
                            int rows, int cols, int trans) {
  int idx = blockIdx.x * blockDim.x + threadIdx.x;
  if (idx >= rows * cols) return;
  int r = idx / cols, c = idx % cols;
  float v = trans ? src[(long)c * rows + r] : src[idx];
  dst[idx] = (_Float16)v;
}

// ---------------------------------------------------------------------------
// im2col for 3x3 stride1 pad1:  out[b][n][k], k = c*9 + ky*3 + kx  (f16)
// in: [b][Cin][H][W] f32
// ---------------------------------------------------------------------------
__global__ void k_im2col3x3(const float* __restrict__ in, _Float16* __restrict__ out,
                            int Cin, int H, int W, long total) {
  long idx = (long)blockIdx.x * blockDim.x + threadIdx.x;
  if (idx >= total) return;
  int K = Cin * 9;
  int k = (int)(idx % K);
  long rem = idx / K;
  int n = (int)(rem % (H * W));
  int b = (int)(rem / (H * W));
  int c = k / 9, t = k % 9;
  int ky = t / 3 - 1, kx = t % 3 - 1;
  int oh = n / W, ow = n % W;
  int ih = oh + ky, iw = ow + kx;
  float v = 0.f;
  if (ih >= 0 && ih < H && iw >= 0 && iw < W)
    v = in[(((long)b * Cin + c) * H + ih) * W + iw];
  out[idx] = (_Float16)v;
}

// ---------------------------------------------------------------------------
// Build value-projection B matrix: cat16[b][n][k], k<128 -> feat0(+lvl), k>=128 -> feat1
// ---------------------------------------------------------------------------
__global__ void k_build_cat(const float* __restrict__ f0l0, const float* __restrict__ f0l1,
                            const float* __restrict__ f1l0, const float* __restrict__ f1l1,
                            const float* __restrict__ lvl, _Float16* __restrict__ out,
                            long total) {
  long idx = (long)blockIdx.x * blockDim.x + threadIdx.x;
  if (idx >= total) return;
  int k = (int)(idx % 256);
  long rem = idx / 256;
  int n = (int)(rem % NTOT);
  int b = (int)(rem / NTOT);
  int c = k & 127;
  const float* pl0 = (k >= 128) ? f1l0 : f0l0;
  const float* pl1 = (k >= 128) ? f1l1 : f0l1;
  float v;
  if (n < N0) v = pl0[((long)b * CCH + c) * N0 + n] + lvl[c];
  else        v = pl1[((long)b * CCH + c) * N1 + (n - N0)] + lvl[CCH + c];
  out[idx] = (_Float16)v;
}

// ---------------------------------------------------------------------------
// WMMA GEMM:  D[m][n] = sum_k A[m][k]*Bt[n][k]  (+bias[m], optional lrelu)
//   A  : f16 row-major  M x K      (weights, shared across batch)
//   Bt : f16 row-major  N x K      (per batch, stride strideB elements)
//   mode 0: C[b][m][n] (M-major)   mode 1: C[b][n][m] (N-major), optional f16 copy C16
// One wave per block; each wave computes a 32(M) x 64(N) tile:
//   8 WMMAs per k-step from 4 A-loads + 8 B-loads (b128 each).
// Requires M % 32 == 0, N % 64 == 0, K % 32 == 0.
// ---------------------------------------------------------------------------
__global__ __launch_bounds__(32) void k_gemm_wmma(
    const _Float16* __restrict__ A, const _Float16* __restrict__ Bt,
    const float* __restrict__ bias, float* __restrict__ C, _Float16* __restrict__ C16,
    int M, int N, int K, long strideB, long strideC, int mode, int act) {
  const int lane = threadIdx.x;
  const int half = lane >> 4;
  const int l16  = lane & 15;
  const int n0 = blockIdx.x * 64;
  const int m0 = blockIdx.y * 32;
  const int b  = blockIdx.z;
  const _Float16* Bb = Bt + (long)b * strideB;

  v8f acc[2][4];
#pragma unroll
  for (int mt = 0; mt < 2; ++mt)
#pragma unroll
    for (int t = 0; t < 4; ++t) acc[mt][t] = (v8f){};

  for (int k0 = 0; k0 < K; k0 += 32) {
    // --- A fragments (two 16x32 tiles): lane holds row m0+mt*16+l16,
    //     K = half*16..+16 contiguous (elems 0..15)
    v16h af[2];
#pragma unroll
    for (int mt = 0; mt < 2; ++mt) {
      const _Float16* pa = A + (long)(m0 + mt * 16 + l16) * K + k0 + half * 16;
      v8h alo = *(const v8h*)(pa);
      v8h ahi = *(const v8h*)(pa + 8);
#pragma unroll
      for (int i = 0; i < 8; ++i) { af[mt][i] = alo[i]; af[mt][i + 8] = ahi[i]; }
    }

    // --- B fragments (four 32x16 tiles): lane holds col n0+16t+l16,
    //     K = half*16..+16 contiguous (one 32B load)
    const _Float16* pb = Bb + (long)(n0 + l16) * K + k0 + half * 16;
    v16h bf[4];
#pragma unroll
    for (int t = 0; t < 4; ++t) bf[t] = *(const v16h*)(pb + (long)(16 * t) * K);

#pragma unroll
    for (int mt = 0; mt < 2; ++mt) {
#pragma unroll
      for (int t = 0; t < 4; ++t) {
        acc[mt][t] = __builtin_amdgcn_wmma_f32_16x16x32_f16(
            false, af[mt], false, bf[t], (short)0, acc[mt][t], false, false);
      }
    }
  }

  // --- epilogue: VGPR r holds row m0 + mt*16 + r + 8*half, col n0+16t+l16
  float* Cb = C + (long)b * strideC;
#pragma unroll
  for (int mt = 0; mt < 2; ++mt) {
#pragma unroll
    for (int t = 0; t < 4; ++t) {
      int col = n0 + t * 16 + l16;
#pragma unroll
      for (int r = 0; r < 8; ++r) {
        int row = m0 + mt * 16 + r + 8 * half;
        float v = acc[mt][t][r] + (bias ? bias[row] : 0.f);
        if (act) v = (v >= 0.f) ? v : 0.1f * v;
        if (mode == 0) {
          Cb[(long)row * N + col] = v;
        } else {
          Cb[(long)col * M + row] = v;
          if (C16) C16[(long)b * N * M + (long)col * M + row] = (_Float16)v;
        }
      }
    }
  }
}

// ---------------------------------------------------------------------------
// Sine positional encoding + NCHW->NHWC transpose; writes f32 and f16 query
// act3: [b][128][1536];  q[b][n][d] = act3[b][d][n] + pos(d, n)
// ---------------------------------------------------------------------------
__global__ void k_posenc(const float* __restrict__ act3, float* __restrict__ qf,
                         _Float16* __restrict__ qh, long total) {
  long idx = (long)blockIdx.x * blockDim.x + threadIdx.x;
  if (idx >= total) return;
  int d = (int)(idx % DCH);
  long rem = idx / DCH;
  int n = (int)(rem % N1);
  int b = (int)(rem / N1);
  int hh = n / W1, ww = n % W1;
  const float TWO_PI = 6.283185307179586f;
  int c = (d < 64) ? d : d - 64;
  float coord = (d < 64) ? (float)(hh + 1) * (TWO_PI / (32.0f + 1e-6f))
                         : (float)(ww + 1) * (TWO_PI / (48.0f + 1e-6f));
  float dim = powf(10000.f, (float)(2 * (c / 2)) / 64.f);
  float ang = coord / dim;
  float pos = (c & 1) ? cosf(ang) : sinf(ang);
  float v = act3[((long)b * DCH + d) * N1 + n] + pos;
  qf[idx] = v;
  qh[idx] = (_Float16)v;
}

// ---------------------------------------------------------------------------
// Softmax over groups of 16 (per b, query, head), in place
// ---------------------------------------------------------------------------
__global__ void k_softmax16(float* __restrict__ aw, long groups) {
  long g = (long)blockIdx.x * blockDim.x + threadIdx.x;
  if (g >= groups) return;
  float* p = aw + g * 16;
  float m = p[0];
#pragma unroll
  for (int i = 1; i < 16; ++i) m = fmaxf(m, p[i]);
  float s = 0.f;
  float e[16];
#pragma unroll
  for (int i = 0; i < 16; ++i) { e[i] = expf(p[i] - m); s += e[i]; }
  float inv = 1.f / s;
#pragma unroll
  for (int i = 0; i < 16; ++i) p[i] = e[i] * inv;
}

// ---------------------------------------------------------------------------
// Multi-scale deformable sampling.
//   value: [b][7680][128] f32 (shared across heads)
//   off  : [b][n][128]  (h,l,p,2)     aw: [b][n][64] (h,l,p) softmaxed
//   out  : f16 [b][n][512]   (k = h*128 + d)  -> B matrix for op GEMM
// block (128,4): x=d, y=h ; grid (1536, 4)
// ---------------------------------------------------------------------------
__global__ __launch_bounds__(512) void k_msda(const float* __restrict__ value,
                                              const float* __restrict__ off,
                                              const float* __restrict__ aw,
                                              _Float16* __restrict__ out) {
  const int d = threadIdx.x;
  const int h = threadIdx.y;
  const int n = blockIdx.x;
  const int b = blockIdx.y;
  const int hh = n / W1, ww = n % W1;
  const float refx = (ww + 0.5f) / (float)W1;
  const float refy = (hh + 0.5f) / (float)H1;
  const float* offp = off + ((long)b * N1 + n) * 128;
  const float* awp  = aw  + ((long)b * N1 + n) * 64;
  const float* vb   = value + (long)b * NTOT * DCH;

  float acc = 0.f;
#pragma unroll
  for (int l = 0; l < NLV; ++l) {
    const int Hl = l ? H1 : H0;
    const int Wl = l ? W1 : W0;
    const int st = l ? N0 : 0;
#pragma unroll
    for (int p = 0; p < NPT; ++p) {
      int j = (h * NLV + l) * NPT + p;
      float lx = refx + offp[2 * j]     / (float)Wl;
      float ly = refy + offp[2 * j + 1] / (float)Hl;
      float x = lx * Wl - 0.5f;
      float y = ly * Hl - 0.5f;
      float x0f = floorf(x), y0f = floorf(y);
      float fx = x - x0f, fy = y - y0f;
      int x0 = (int)x0f, y0 = (int)y0f;
      float s = 0.f;
      {
        int xi = x0, yi = y0;
        if (xi >= 0 && xi < Wl && yi >= 0 && yi < Hl)
          s += (1.f - fx) * (1.f - fy) * vb[(long)(st + yi * Wl + xi) * DCH + d];
      }
      {
        int xi = x0 + 1, yi = y0;
        if (xi >= 0 && xi < Wl && yi >= 0 && yi < Hl)
          s += fx * (1.f - fy) * vb[(long)(st + yi * Wl + xi) * DCH + d];
      }
      {
        int xi = x0, yi = y0 + 1;
        if (xi >= 0 && xi < Wl && yi >= 0 && yi < Hl)
          s += (1.f - fx) * fy * vb[(long)(st + yi * Wl + xi) * DCH + d];
      }
      {
        int xi = x0 + 1, yi = y0 + 1;
        if (xi >= 0 && xi < Wl && yi >= 0 && yi < Hl)
          s += fx * fy * vb[(long)(st + yi * Wl + xi) * DCH + d];
      }
      acc += awp[h * 16 + l * NPT + p] * s;
    }
  }
  out[((long)b * N1 + n) * (NHD * DCH) + h * DCH + d] = (_Float16)acc;
}

// ---------------------------------------------------------------------------
// Host-side launcher
// ---------------------------------------------------------------------------
static inline long cdivl(long a, long b) { return (a + b - 1) / b; }

extern "C" void kernel_launch(void* const* d_in, const int* in_sizes, int n_in,
                              void* d_out, int out_size, void* d_ws, size_t ws_size,
                              hipStream_t stream) {
  // ---- inputs (setup_inputs order), all f32 ----
  const float* feat0_l0 = (const float*)d_in[0];
  const float* feat0_l1 = (const float*)d_in[1];
  const float* feat1_l0 = (const float*)d_in[2];
  const float* feat1_l1 = (const float*)d_in[3];
  const float* cw1 = (const float*)d_in[4];   const float* cb1 = (const float*)d_in[5];
  const float* cw2 = (const float*)d_in[6];   const float* cb2 = (const float*)d_in[7];
  const float* cw3 = (const float*)d_in[8];   const float* cb3 = (const float*)d_in[9];
  const float* vw  = (const float*)d_in[10];  const float* vb  = (const float*)d_in[11];
  const float* lvl = (const float*)d_in[12];
  const float* so_w[2] = {(const float*)d_in[13], (const float*)d_in[19]};
  const float* so_b[2] = {(const float*)d_in[14], (const float*)d_in[20]};
  const float* aw_w[2] = {(const float*)d_in[15], (const float*)d_in[21]};
  const float* aw_b[2] = {(const float*)d_in[16], (const float*)d_in[22]};
  const float* op_w[2] = {(const float*)d_in[17], (const float*)d_in[23]};
  const float* op_b[2] = {(const float*)d_in[18], (const float*)d_in[24]};
  float* out = (float*)d_out;

  // ---- workspace carve-out (256B aligned) ----
  char* ws = (char*)d_ws;
  size_t off_b = 0;
  auto alloc = [&](size_t bytes) -> void* {
    void* p = ws + off_b;
    off_b = (off_b + bytes + 255) & ~(size_t)255;
    return p;
  };
  _Float16* w_cw1 = (_Float16*)alloc((size_t)256 * 1152 * 2);
  _Float16* w_cw2 = (_Float16*)alloc((size_t)128 * 2304 * 2);
  _Float16* w_cw3 = (_Float16*)alloc((size_t)128 * 1152 * 2);
  _Float16* w_vw  = (_Float16*)alloc((size_t)128 * 256 * 2);
  _Float16* w_so[2] = {(_Float16*)alloc((size_t)128 * 128 * 2), (_Float16*)alloc((size_t)128 * 128 * 2)};
  _Float16* w_aw[2] = {(_Float16*)alloc((size_t)64 * 128 * 2),  (_Float16*)alloc((size_t)64 * 128 * 2)};
  _Float16* w_op[2] = {(_Float16*)alloc((size_t)128 * 512 * 2), (_Float16*)alloc((size_t)128 * 512 * 2)};
  _Float16* Bbuf   = (_Float16*)alloc((size_t)BSZ * N1 * 2304 * 2);   // largest im2col; reused
  _Float16* qf16   = (_Float16*)alloc((size_t)BSZ * N1 * DCH * 2);
  _Float16* msda16 = (_Float16*)alloc((size_t)BSZ * N1 * NHD * DCH * 2);
  float* valuef = (float*)alloc((size_t)BSZ * NTOT * DCH * 4);
  float* act1   = (float*)alloc((size_t)BSZ * 256 * N1 * 4);
  float* act2   = (float*)alloc((size_t)BSZ * 128 * N1 * 4);
  float* act3   = (float*)alloc((size_t)BSZ * 128 * N1 * 4);
  float* qf32   = (float*)alloc((size_t)BSZ * N1 * DCH * 4);
  float* offbuf = (float*)alloc((size_t)BSZ * N1 * 128 * 4);
  float* awbuf  = (float*)alloc((size_t)BSZ * N1 * 64 * 4);
  (void)ws_size; (void)n_in; (void)in_sizes; (void)out_size;

  const int T = 256;
  auto g1 = [&](long total) { return dim3((unsigned)cdivl(total, T)); };

  // ---- 1) weight conversion (A matrices, row-major M x K, f16) ----
  k_convert_w<<<g1(256L*1152), T, 0, stream>>>(cw1, w_cw1, 256, 1152, 0);
  k_convert_w<<<g1(128L*2304), T, 0, stream>>>(cw2, w_cw2, 128, 2304, 0);
  k_convert_w<<<g1(128L*1152), T, 0, stream>>>(cw3, w_cw3, 128, 1152, 0);
  k_convert_w<<<g1(128L*256),  T, 0, stream>>>(vw,  w_vw,  128, 256,  0);
  for (int i = 0; i < 2; ++i) {
    k_convert_w<<<g1(128L*128), T, 0, stream>>>(so_w[i], w_so[i], 128, 128, 1);
    k_convert_w<<<g1(64L*128),  T, 0, stream>>>(aw_w[i], w_aw[i], 64,  128, 1);
    k_convert_w<<<g1(128L*512), T, 0, stream>>>(op_w[i], w_op[i], 128, 512, 1);
  }

  // ---- 2) value projection: value[b][n][128] = vw @ cat + vb ----
  {
    long tot = (long)BSZ * NTOT * 256;
    k_build_cat<<<g1(tot), T, 0, stream>>>(feat0_l0, feat0_l1, feat1_l0, feat1_l1, lvl, Bbuf, tot);
    dim3 grid(NTOT / 64, 128 / 32, BSZ);
    k_gemm_wmma<<<grid, 32, 0, stream>>>(w_vw, Bbuf, vb, valuef, nullptr,
                                         128, NTOT, 256, (long)NTOT * 256, (long)128 * NTOT, 1, 0);
  }

  // ---- 3) conv stack on feat0_l1 (implicit GEMM, lrelu) ----
  {
    long tot = (long)BSZ * N1 * 1152;
    k_im2col3x3<<<g1(tot), T, 0, stream>>>(feat0_l1, Bbuf, 128, H1, W1, tot);
    dim3 grid(N1 / 64, 256 / 32, BSZ);
    k_gemm_wmma<<<grid, 32, 0, stream>>>(w_cw1, Bbuf, cb1, act1, nullptr,
                                         256, N1, 1152, (long)N1 * 1152, (long)256 * N1, 0, 1);
  }
  {
    long tot = (long)BSZ * N1 * 2304;
    k_im2col3x3<<<g1(tot), T, 0, stream>>>(act1, Bbuf, 256, H1, W1, tot);
    dim3 grid(N1 / 64, 128 / 32, BSZ);
    k_gemm_wmma<<<grid, 32, 0, stream>>>(w_cw2, Bbuf, cb2, act2, nullptr,
                                         128, N1, 2304, (long)N1 * 2304, (long)128 * N1, 0, 1);
  }
  {
    long tot = (long)BSZ * N1 * 1152;
    k_im2col3x3<<<g1(tot), T, 0, stream>>>(act2, Bbuf, 128, H1, W1, tot);
    dim3 grid(N1 / 64, 128 / 32, BSZ);
    k_gemm_wmma<<<grid, 32, 0, stream>>>(w_cw3, Bbuf, cb3, act3, nullptr,
                                         128, N1, 1152, (long)N1 * 1152, (long)128 * N1, 0, 1);
  }

  // ---- 4) positional encoding + transpose to query [b][n][d] (f32 + f16) ----
  {
    long tot = (long)BSZ * N1 * DCH;
    k_posenc<<<g1(tot), T, 0, stream>>>(act3, qf32, qf16, tot);
  }

  // ---- 5) two deformable-attention blocks ----
  for (int i = 0; i < 2; ++i) {
    // sampling offsets: off[b][n][128]
    {
      dim3 grid(N1 / 64, 128 / 32, BSZ);
      k_gemm_wmma<<<grid, 32, 0, stream>>>(w_so[i], qf16, so_b[i], offbuf, nullptr,
                                           128, N1, 128, (long)N1 * 128, (long)128 * N1, 1, 0);
    }
    // attention logits: aw[b][n][64]
    {
      dim3 grid(N1 / 64, 64 / 32, BSZ);
      k_gemm_wmma<<<grid, 32, 0, stream>>>(w_aw[i], qf16, aw_b[i], awbuf, nullptr,
                                           64, N1, 128, (long)N1 * 128, (long)64 * N1, 1, 0);
    }
    // softmax over (NL*NP)=16 per head
    {
      long groups = (long)BSZ * N1 * NHD;
      k_softmax16<<<g1(groups), T, 0, stream>>>(awbuf, groups);
    }
    // deformable sampling -> msda16 [b][n][512]
    {
      dim3 blk(DCH, NHD);
      dim3 grid(N1, BSZ);
      k_msda<<<grid, blk, 0, stream>>>(valuef, offbuf, awbuf, msda16);
    }
    // output projection
    {
      dim3 grid(N1 / 64, 128 / 32, BSZ);
      if (i == 0) {
        // next block's query: [b][n][128] f32 + f16
        k_gemm_wmma<<<grid, 32, 0, stream>>>(w_op[i], msda16, op_b[i], qf32, qf16,
                                             128, N1, 512, (long)N1 * 512, (long)128 * N1, 1, 0);
      } else {
        // final output: [b][d][h][w] == M-major [b][128][1536]
        k_gemm_wmma<<<grid, 32, 0, stream>>>(w_op[i], msda16, op_b[i], out, nullptr,
                                             128, N1, 512, (long)N1 * 512, (long)128 * N1, 0, 0);
      }
    }
  }
}
